// BondPoolingLayer_52716428591258
// MI455X (gfx1250) — matrix operand order repaired and checked by reading, not compiled, not run on hardware
//
#include <hip/hip_runtime.h>

// CDNA5 / gfx1250 bond-MLP kernel.
// Strategy: 16-edge tiles per wave32, bf16 WMMA (v_wmma_f32_16x16x32_bf16),
// weights packed in LDS in B-operand layout, per-wave LDS scratch for
// layer->layer relayout (D layout -> A layout).

typedef __attribute__((ext_vector_type(16))) __bf16 v16bf;
typedef __attribute__((ext_vector_type(8)))  __bf16 v8bf;
typedef __attribute__((ext_vector_type(8)))  float  v8f;

#define DQ      128       // feature dim
#define BLOCK   128       // threads per block = 4 waves
#define WAVES   4

// LDS byte offsets
#define W1_OFF   0                    // 8 kb-tiles x 8 nt-tiles x 1024B = 64KB
#define W2_OFF   65536                // 4 x 8 x 1024B = 32KB
#define W3_OFF   98304                // 4 x 1 x 1024B = 4KB (N padded 2->16)
#define SCR_OFF  102400               // per-wave: scrF(4KB) + scrR(4KB)
#define SMEM_BYTES (SCR_OFF + WAVES * 8192)   // 135168 B

// Single-instruction ReLU. fmaxf()/fmed3 builtins lower to a canonicalize
// v_max + v_max pair on gfx1250; force the 1-op form via inline asm
// (gfx1250 uses the *_num opcode naming): med3(x, 0, +inf) == max(x, 0).
__device__ __forceinline__ float relu1(float x) {
    float r;
    asm("v_med3_num_f32 %0, %1, 0, 0x7f800000" : "=v"(r) : "v"(x));
    return r;
}

__device__ __forceinline__ v8f wmma_bf16(v16bf a, v16bf b, v8f c) {
    // 8 args: (neg_a, A, neg_b, B, c_mod, C, reuse_a, reuse_b)
    return __builtin_amdgcn_wmma_f32_16x16x32_bf16(
        false, a, false, b, (short)0, c, false, false);
}

// B-operand tile load from packed LDS: lane's 16 bf16 are contiguous (32B).
__device__ __forceinline__ v16bf load_b_tile(const __bf16* base, int lane) {
    return *(const v16bf*)(base + lane * 16);
}

// A-operand build from a global fp32 row.
// Lane group 0 (lanes 0-15):  elements = row[ko..ko+7], row[ko+16..ko+23]
// Lane group 1 (lanes 16-31): caller passes ko with +8 (K {8..15,24..31}).
__device__ __forceinline__ v16bf load_a_global(const float* __restrict__ row, int ko) {
    const float* p0 = row + ko;
    const float* p1 = row + ko + 16;
    v16bf a;
#pragma unroll
    for (int i = 0; i < 8; ++i) { a[i] = (__bf16)p0[i]; a[8 + i] = (__bf16)p1[i]; }
    return a;
}

// Same as above but from a bf16 LDS row (two 16B chunks -> ds_load_b128).
__device__ __forceinline__ v16bf load_a_lds(const __bf16* row, int ko) {
    v8bf lo = *(const v8bf*)(row + ko);
    v8bf hi = *(const v8bf*)(row + ko + 16);
    return __builtin_shufflevector(lo, hi, 0,1,2,3,4,5,6,7,8,9,10,11,12,13,14,15);
}

// Pack fp32 row-major KxN (N==128) weight into WMMA B-operand tile layout:
// tile (kb, nt) is 512 bf16; slot s (==lane) holds 16 K values
// for column N = nt*16 + (s&15), K = kb*32 + {0..7,16..23} + (s>=16 ? 8 : 0).
__device__ __forceinline__ void pack_weight(const float* __restrict__ W, __bf16* dstp,
                                            int K, int ntiles_n, int tid) {
    const int total = K * 128;
    for (int i = tid; i < total; i += BLOCK) {
        int k = i >> 7, n = i & 127;
        int kb = k >> 5, kin = k & 31, nt = n >> 4, nin = n & 15;
        int idx = ((kb * ntiles_n + nt) << 9) + (((((kin >> 3) & 1) << 4) + nin) << 4) + (kin & 7) + ((kin >> 4) << 3);
        dstp[idx] = (__bf16)W[i];
    }
}

__global__ void __launch_bounds__(BLOCK)
bond_mlp_kernel(const float* __restrict__ h,
                const int*   __restrict__ src,
                const int*   __restrict__ dst,
                const float* __restrict__ W1, const float* __restrict__ b1,
                const float* __restrict__ W2, const float* __restrict__ b2,
                const float* __restrict__ W3, const float* __restrict__ b3,
                float* __restrict__ out, int n_edges, int n_tiles)
{
    extern __shared__ __align__(16) char smem[];
    __bf16* w1p = (__bf16*)(smem + W1_OFF);
    __bf16* w2p = (__bf16*)(smem + W2_OFF);
    __bf16* w3p = (__bf16*)(smem + W3_OFF);

    const int tid = threadIdx.x;

    // ---- stage weights into LDS (bf16, B-operand layout), once per block ----
    pack_weight(W1, w1p, 256, 8, tid);   // 8 kb-tiles (K=256), 8 nt-tiles
    pack_weight(W2, w2p, 128, 8, tid);   // 4 kb-tiles, 8 nt-tiles
    for (int i = tid; i < 128 * 16; i += BLOCK) {   // W3 padded N=2 -> 16
        int k = i >> 4, n = i & 15;
        float v = (n < 2) ? W3[k * 2 + n] : 0.0f;
        int kb = k >> 5, kin = k & 31;
        int idx = (kb << 9) + (((((kin >> 3) & 1) << 4) + n) << 4) + (kin & 7) + ((kin >> 4) << 3);
        w3p[idx] = (__bf16)v;
    }
    __syncthreads();

    const int wave = tid >> 5;
    const int lane = tid & 31;
    const int m    = lane & 15;      // row within tile (A) / column N (B,C,D)
    const int hig  = lane >> 4;      // lane half-group
    const int ksub = hig << 3;       // A-operand K sub-offset: 0 or 8

    __bf16* scrF = (__bf16*)(smem + SCR_OFF + wave * 8192);  // 16x128 bf16
    __bf16* scrR = scrF + 2048;                              // 16x128 bf16

    const float b3v = (m < 2) ? 2.0f * b3[m] : 0.0f;

    const int gwave  = blockIdx.x * WAVES + wave;
    const int nwaves = gridDim.x * WAVES;

    for (int t = gwave; t < n_tiles; t += nwaves) {
        const int e0 = t << 4;
        int er = e0 + m; if (er >= n_edges) er = n_edges - 1;
        const float* rowS = h + (size_t)src[er] * DQ;
        const float* rowD = h + (size_t)dst[er] * DQ;

        // ---- gather A operands for both endpoints (4 K-blocks each) ----
        v16bf As[4], Ad[4];
#pragma unroll
        for (int kb = 0; kb < 4; ++kb) {
            int ko = (kb << 5) + ksub;
            As[kb] = load_a_global(rowS, ko);
            Ad[kb] = load_a_global(rowD, ko);
        }

        // ---- layer 1: [16,256]@[256,128] for forward + reverse ----
#pragma unroll
        for (int nt = 0; nt < 8; ++nt) {
            float bias = b1[(nt << 4) + m];
            v8f accF, accR;
#pragma unroll
            for (int r = 0; r < 8; ++r) { accF[r] = bias; accR[r] = bias; }
#pragma unroll
            for (int kb = 0; kb < 4; ++kb) {
                v16bf Ba = load_b_tile(w1p + (((kb    ) * 8 + nt) << 9), lane); // W1 rows 0..127
                v16bf Bb = load_b_tile(w1p + (((kb + 4) * 8 + nt) << 9), lane); // W1 rows 128..255
                accF = wmma_bf16(As[kb], Ba, accF);
                accF = wmma_bf16(Ad[kb], Bb, accF);
                accR = wmma_bf16(Ad[kb], Ba, accR);
                accR = wmma_bf16(As[kb], Bb, accR);
            }
#pragma unroll
            for (int r = 0; r < 8; ++r) {   // D layout: M = r + 8*hig, N = m
                int mm = r + (hig << 3);
                scrF[mm * DQ + (nt << 4) + m] = (__bf16)relu1(accF[r]);
                scrR[mm * DQ + (nt << 4) + m] = (__bf16)relu1(accR[r]);
            }
        }

        // ---- layer 2: [16,128]@[128,128], both directions ----
        v16bf Af[4], Ar[4];
#pragma unroll
        for (int kb = 0; kb < 4; ++kb) {
            int ko = (kb << 5) + ksub;
            Af[kb] = load_a_lds(scrF + m * DQ, ko);
            Ar[kb] = load_a_lds(scrR + m * DQ, ko);
        }
#pragma unroll
        for (int nt = 0; nt < 8; ++nt) {
            float bias = b2[(nt << 4) + m];
            v8f accF, accR;
#pragma unroll
            for (int r = 0; r < 8; ++r) { accF[r] = bias; accR[r] = bias; }
#pragma unroll
            for (int kb = 0; kb < 4; ++kb) {
                v16bf B = load_b_tile(w2p + ((kb * 8 + nt) << 9), lane);
                accF = wmma_bf16(Af[kb], B, accF);
                accR = wmma_bf16(Ar[kb], B, accR);
            }
#pragma unroll
            for (int r = 0; r < 8; ++r) {
                int mm = r + (hig << 3);
                float s = relu1(accF[r]) + relu1(accR[r]);   // relu(L2f)+relu(L2r)
                scrF[mm * DQ + (nt << 4) + m] = (__bf16)s;
            }
        }

        // ---- layer 3: (L2f+L2r) @ W3 + 2*b3 ----
        v16bf A3[4];
#pragma unroll
        for (int kb = 0; kb < 4; ++kb)
            A3[kb] = load_a_lds(scrF + m * DQ, (kb << 5) + ksub);
        v8f acc3;
#pragma unroll
        for (int r = 0; r < 8; ++r) acc3[r] = b3v;
#pragma unroll
        for (int kb = 0; kb < 4; ++kb)
            acc3 = wmma_bf16(A3[kb], load_b_tile(w3p + (kb << 9), lane), acc3);

        if (m < 2) {
#pragma unroll
            for (int r = 0; r < 8; ++r) {
                int e = e0 + r + (hig << 3);
                if (e < n_edges) out[e * 2 + m] = acc3[r];
            }
        }
    }
}

extern "C" void kernel_launch(void* const* d_in, const int* in_sizes, int n_in,
                              void* d_out, int out_size, void* d_ws, size_t ws_size,
                              hipStream_t stream) {
    const float* h   = (const float*)d_in[0];
    const int*   src = (const int*)d_in[1];   // jnp int64 w/o x64 -> int32 per harness contract
    const int*   dst = (const int*)d_in[2];
    const float* W1  = (const float*)d_in[3];
    const float* b1  = (const float*)d_in[4];
    const float* W2  = (const float*)d_in[5];
    const float* b2  = (const float*)d_in[6];
    const float* W3  = (const float*)d_in[7];
    const float* b3  = (const float*)d_in[8];
    float* out = (float*)d_out;

    const int n_edges = in_sizes[1];
    const int n_tiles = (n_edges + 15) / 16;
    int grid = (n_tiles + WAVES - 1) / WAVES;
    if (grid > 688) grid = 688;   // amortize LDS weight packing; grid-stride over tiles
    if (grid < 1)   grid = 1;

    bond_mlp_kernel<<<grid, BLOCK, SMEM_BYTES, stream>>>(
        h, src, dst, W1, b1, W2, b2, W3, b3, out, n_edges, n_tiles);
}
